// DynamicConv_670014898566
// MI455X (gfx1250) — matrix operand order repaired and checked by reading, not compiled, not run on hardware
//
#include <hip/hip_runtime.h>

#define B_    32
#define CIN   64
#define H_    128
#define W_    128
#define COUT  64
#define E_    4
#define KTOT  576        // CIN * 9
#define KSTEPS 18        // KTOT / 32
#define CPITCH 72        // channel pitch (elements) in slabT: 64 + 8 pad (144 B, conflict-free)
#define COLS   130       // W_ + 2 guard columns

typedef __bf16 bf16_t;
typedef bf16_t v16bf __attribute__((ext_vector_type(16)));
typedef float  v8f   __attribute__((ext_vector_type(8)));
typedef unsigned short u16x8 __attribute__((ext_vector_type(8)));

union BF16Frag { v16bf v; u16x8 h[2]; };

__device__ __forceinline__ unsigned short f2bf(float f) {
  return __builtin_bit_cast(unsigned short, (bf16_t)f);   // hw cvt, RNE
}

// ---------------- Kernel 1: global average pool -> pooled[B*CIN] ----------------
__global__ void dynconv_pool_kernel(const float* __restrict__ x,
                                    float* __restrict__ pooled) {
  __shared__ float red[256];
  const int bc = blockIdx.x;                 // b*CIN + c
  const int t  = threadIdx.x;
  const float4* src = (const float4*)(x + (size_t)bc * (H_ * W_));
  float s = 0.f;
#pragma unroll
  for (int i = 0; i < 16; ++i) {             // 16384/4/256 = 16 float4 per thread
    float4 v = src[t + i * 256];
    s += v.x + v.y + v.z + v.w;
  }
  red[t] = s;
  __syncthreads();
  for (int off = 128; off > 0; off >>= 1) {
    if (t < off) red[t] += red[t + off];
    __syncthreads();
  }
  if (t == 0) pooled[bc] = red[0] * (1.0f / (float)(H_ * W_));
}

// ---------------- Kernel 2: routing = sigmoid(pooled @ w_route^T + b) -----------
__global__ void dynconv_route_kernel(const float* __restrict__ pooled,
                                     const float* __restrict__ w_route,
                                     const float* __restrict__ b_route,
                                     float* __restrict__ routing) {
  const int t = threadIdx.x;
  if (t < B_ * E_) {
    const int b = t >> 2, e = t & 3;
    float acc = b_route[e];
#pragma unroll 8
    for (int c = 0; c < CIN; ++c)
      acc += pooled[b * CIN + c] * w_route[e * CIN + c];
    routing[t] = 1.0f / (1.0f + __expf(-acc));
  }
}

// ---- Kernel 3: mixed weights, f32 -> bf16, K-permuted to k' = p*64 + i ---------
// wmix[b][o][p*64 + i] = sum_e routing[b,e] * w_experts[e][o][i][p]   (p = kh*3+kw)
__global__ void dynconv_mix_kernel(const float* __restrict__ w_experts,
                                   const float* __restrict__ routing,
                                   unsigned short* __restrict__ wmix) {
  const int b = blockIdx.x;
  const float r0 = routing[b * 4 + 0];
  const float r1 = routing[b * 4 + 1];
  const float r2 = routing[b * 4 + 2];
  const float r3 = routing[b * 4 + 3];
  const int n = COUT * KTOT;                 // 36864
  for (int dst = threadIdx.x; dst < n; dst += 256) {
    int o   = dst / KTOT;
    int rem = dst - o * KTOT;
    int p   = rem >> 6;                      // kernel position 0..8
    int i   = rem & 63;                      // input channel
    int src = o * KTOT + i * 9 + p;          // expert layout [o][i][kh][kw]
    float v = r0 * w_experts[src] +
              r1 * w_experts[n + src] +
              r2 * w_experts[2 * n + src] +
              r3 * w_experts[3 * n + src];
    wmix[(size_t)b * n + dst] = f2bf(v);
  }
}

// ---------------- Kernel 4: implicit-GEMM conv via WMMA bf16 --------------------
// grid = (H, B), block = 256 (8 waves). Block computes y[b, 0:64, h, 0:128].
// K order: k' = (kh*3+kw)*64 + i  ->  each K-step of 32 has fixed (kh,kw).
// Wave tile: 32(M) x 32(N): 2 A-frags x 2 B-frags -> 4 WMMA, 1 ds_b128/WMMA.
__global__ __launch_bounds__(256) void dynconv_conv_kernel(
    const float* __restrict__ x,
    const unsigned short* __restrict__ wmix,
    float* __restrict__ y) {
  // slabT[kh][col][channel]: col 0 and 129 are zero guards (SAME padding in w)
  __shared__ __align__(16) unsigned short slabT[3 * COLS * CPITCH];  // 56160 B

  const int h = blockIdx.x;
  const int b = blockIdx.y;
  const int t = threadIdx.x;

  // ---- guard columns ----
  if (t < 192) {
    int kh = t >> 6, c = t & 63;
    slabT[(kh * COLS + 0) * CPITCH + c]   = 0;
    slabT[(kh * COLS + 129) * CPITCH + c] = 0;
  }
  // ---- main fill: x[b, c, h+kh-1, w] -> slabT[kh][w+1][c] (bf16) ----
  {
    const float* xb = x + (size_t)b * CIN * H_ * W_;
    const int w    = t & 127;                // lanes consecutive in w -> coalesced
    const int half = t >> 7;                 // channel half
#pragma unroll
    for (int kh = 0; kh < 3; ++kh) {
      const int hh = h + kh - 1;
      const bool inb = (hh >= 0) && (hh < H_);
      const float* src = xb + (size_t)(inb ? hh : 0) * W_ + w;
#pragma unroll
      for (int cb = 0; cb < 4; ++cb) {
        const int c0 = half * 32 + cb * 8;
        u16x8 v = {};
        if (inb) {
#pragma unroll
          for (int j = 0; j < 8; ++j)
            v[j] = f2bf(src[(size_t)(c0 + j) * (H_ * W_)]);
        }
        *(u16x8*)&slabT[(kh * COLS + w + 1) * CPITCH + c0] = v;
      }
    }
  }
  __syncthreads();

  const int lane = t & 31;
  const int wave = t >> 5;
  const int mi   = wave & 1;                 // 32-row out-channel tile
  const int nj   = wave >> 1;                // 32-col pixel tile (0..3)
  const int lm   = lane & 15;
  const int lhi  = lane >> 4;

  v8f acc00 = {}, acc01 = {}, acc10 = {}, acc11 = {};

  // A-fragment source rows in permuted weights (global, L2/WGP$-resident)
  const unsigned short* wrow0 =
      wmix + (size_t)b * COUT * KTOT + (mi * 32 + lm) * KTOT + lhi * 8;
  const unsigned short* wrow1 = wrow0 + 16 * KTOT;

  // Per-B-fragment base offsets into slabT (element units); per-step delta is imm
  const int bbase0 = (nj * 32 + lm) * CPITCH + lhi * 8;
  const int bbase1 = bbase0 + 16 * CPITCH;

#pragma unroll
  for (int ks = 0; ks < KSTEPS; ++ks) {
    const int p  = ks >> 1;                  // kernel position (kh,kw) fixed per step
    const int i0 = (ks & 1) * 32;            // channel block base
    const int kh = p / 3;
    const int kw = p - kh * 3;
    const int soff = (kh * COLS + kw) * CPITCH + i0;   // compile-time constant

    // A (16x32 bf16): elems 0..7 -> k'=ks*32+lhi*8+0..7 ; 8..15 -> +16
    BF16Frag fa0, fa1;
    fa0.h[0] = *(const u16x8*)(wrow0 + ks * 32);
    fa0.h[1] = *(const u16x8*)(wrow0 + ks * 32 + 16);
    fa1.h[0] = *(const u16x8*)(wrow1 + ks * 32);
    fa1.h[1] = *(const u16x8*)(wrow1 + ks * 32 + 16);

    // B (32x16 bf16): lane lm = pixel col, elems 0..7 -> channels i0+lhi*8+0..7 ; 8..15 -> +16
    BF16Frag fb0, fb1;
    {
      const unsigned short* bp0 = &slabT[bbase0 + soff];
      fb0.h[0] = *(const u16x8*)bp0;
      fb0.h[1] = *(const u16x8*)(bp0 + 16);
      const unsigned short* bp1 = &slabT[bbase1 + soff];
      fb1.h[0] = *(const u16x8*)bp1;
      fb1.h[1] = *(const u16x8*)(bp1 + 16);
    }

    acc00 = __builtin_amdgcn_wmma_f32_16x16x32_bf16(
                false, fa0.v, false, fb0.v, (short)0, acc00, false, false);
    acc01 = __builtin_amdgcn_wmma_f32_16x16x32_bf16(
                false, fa0.v, false, fb1.v, (short)0, acc01, false, false);
    acc10 = __builtin_amdgcn_wmma_f32_16x16x32_bf16(
                false, fa1.v, false, fb0.v, (short)0, acc10, false, false);
    acc11 = __builtin_amdgcn_wmma_f32_16x16x32_bf16(
                false, fa1.v, false, fb1.v, (short)0, acc11, false, false);
  }

  // ---- Store: D layout -> VGPR r holds (M = r + 8*lhi, N = lm) ----
  float* yb = y + (((size_t)b * COUT) * H_ + h) * W_;
#pragma unroll
  for (int mm = 0; mm < 2; ++mm) {
#pragma unroll
    for (int nn = 0; nn < 2; ++nn) {
      const int wcol = nj * 32 + nn * 16 + lm;
      const v8f acc = (mm == 0) ? (nn == 0 ? acc00 : acc01)
                                : (nn == 0 ? acc10 : acc11);
#pragma unroll
      for (int r = 0; r < 8; ++r) {
        const int o = mi * 32 + mm * 16 + r + 8 * lhi;
        yb[(size_t)o * H_ * W_ + wcol] = acc[r];
      }
    }
  }
}

extern "C" void kernel_launch(void* const* d_in, const int* in_sizes, int n_in,
                              void* d_out, int out_size, void* d_ws, size_t ws_size,
                              hipStream_t stream) {
  const float* x         = (const float*)d_in[0];
  const float* w_route   = (const float*)d_in[1];
  const float* b_route   = (const float*)d_in[2];
  const float* w_experts = (const float*)d_in[3];
  float* y = (float*)d_out;

  char* ws = (char*)d_ws;
  float* pooled  = (float*)ws;                         // 2048 f32
  float* routing = (float*)(ws + 8192);                // 128 f32
  unsigned short* wmix = (unsigned short*)(ws + 8704); // 32*64*576 bf16 (~2.3 MB)

  dynconv_pool_kernel <<<dim3(B_ * CIN), dim3(256), 0, stream>>>(x, pooled);
  dynconv_route_kernel<<<dim3(1),        dim3(128), 0, stream>>>(pooled, w_route, b_route, routing);
  dynconv_mix_kernel  <<<dim3(B_),       dim3(256), 0, stream>>>(w_experts, routing, wmix);
  dynconv_conv_kernel <<<dim3(H_, B_),   dim3(256), 0, stream>>>(x, wmix, y);
}
